// PrototypeMemory_45887430590517
// MI455X (gfx1250) — matrix operand order repaired and checked by reading, not compiled, not run on hardware
//
#include <hip/hip_runtime.h>
#include <hip/hip_bf16.h>

// Problem constants (reference: B=8192, C=1024, D=768)
#define B_N 8192
#define C_N 1024
#define D_N 768
#define EPS_F 1e-8f
#define MARGIN_F 0.5f

typedef __attribute__((ext_vector_type(16))) __bf16 v16bf;
typedef __attribute__((ext_vector_type(8)))  float  v8f;
typedef __attribute__((ext_vector_type(4)))  unsigned int u32x4;

// ---------------------------------------------------------------------------
// Fragment loaders for v_wmma_f32_16x16x32_bf16 (ISA 7.12.2 layouts).
// A (16x32, row-major source, stride D_N):
//   lane L: row = L&15 ; chunk0 = K[k0+8*(L>>4) .. +7], chunk1 = chunk0 + 16
// B (32x16 = P^T, so column n of B is row n of P, row-major, stride D_N):
//   lane L: col = L&15 ; chunk0 = K[k0+8*(L>>4) .. +7], chunk1 = chunk0 + 16
// ---------------------------------------------------------------------------
__device__ inline v16bf load_a_frag(const __bf16* __restrict__ base, int row,
                                    int k0, int lane) {
    const int koff = (lane >> 4) * 8;
    const __bf16* p = base + (size_t)row * D_N + k0 + koff;
    union { v16bf v; u32x4 u[2]; } f;
    f.u[0] = *reinterpret_cast<const u32x4*>(p);
    f.u[1] = *reinterpret_cast<const u32x4*>(p + 16);
    return f.v;
}

__device__ inline v16bf load_b_frag(const __bf16* __restrict__ base, int col,
                                    int k0, int lane) {
    const int koff = (lane >> 4) * 8;
    const __bf16* p = base + (size_t)col * D_N + k0 + koff;
    union { v16bf v; u32x4 u[2]; } f;
    f.u[0] = *reinterpret_cast<const u32x4*>(p);
    f.u[1] = *reinterpret_cast<const u32x4*>(p + 16);
    return f.v;
}

// ---------------------------------------------------------------------------
// Kernel 1: fp32 -> (bf16 hi, bf16 lo) split + row squared-norm.
// One block per row of length D_N.
// ---------------------------------------------------------------------------
__global__ __launch_bounds__(256) void pm_convert_kernel(
    const float* __restrict__ src, __bf16* __restrict__ hi,
    __bf16* __restrict__ lo, float* __restrict__ norm2) {
    __shared__ float sred[256];
    const int row = blockIdx.x;
    const size_t base = (size_t)row * D_N;
    float ss = 0.0f;
    for (int k = threadIdx.x; k < D_N; k += 256) {
        float x = src[base + k];
        __bf16 h = (__bf16)x;
        __bf16 l = (__bf16)(x - (float)h);
        hi[base + k] = h;
        lo[base + k] = l;
        ss += x * x;
    }
    sred[threadIdx.x] = ss;
    __syncthreads();
    for (int s = 128; s > 0; s >>= 1) {
        if (threadIdx.x < s) sred[threadIdx.x] += sred[threadIdx.x + s];
        __syncthreads();
    }
    if (threadIdx.x == 0) norm2[row] = sred[0];
}

// ---------------------------------------------------------------------------
// Kernel 2: pos_dist[b] = || e_b - proto[label_b] ||  (exact fp32, no eps)
// ---------------------------------------------------------------------------
__global__ __launch_bounds__(256) void pm_pos_kernel(
    const float* __restrict__ emb, const float* __restrict__ proto,
    const int* __restrict__ labels, float* __restrict__ pos_dist) {
    __shared__ float sred[256];
    const int b = blockIdx.x;
    const int lab = labels[b];
    const float* e = emb + (size_t)b * D_N;
    const float* p = proto + (size_t)lab * D_N;
    float ss = 0.0f;
    for (int k = threadIdx.x; k < D_N; k += 256) {
        float d = e[k] - p[k];
        ss += d * d;
    }
    sred[threadIdx.x] = ss;
    __syncthreads();
    for (int s = 128; s > 0; s >>= 1) {
        if (threadIdx.x < s) sred[threadIdx.x] += sred[threadIdx.x + s];
        __syncthreads();
    }
    if (threadIdx.x == 0) pos_dist[b] = sqrtf(sred[0]);
}

// ---------------------------------------------------------------------------
// Kernel 3: fused split-bf16 WMMA GEMM + masked logsumexp per row.
// Grid: B/16 blocks of 256 threads (8 waves). Block owns rows [b0, b0+16);
// wave w owns column tiles [w*8, w*8+8) of 64 -> full C=1024 coverage.
// dot(e_b, p_c) via 3x v_wmma_f32_16x16x32_bf16 per 32-wide K step (24 steps).
// C/D layout: VGPR r, lane L -> M = r + 8*(L>>4), N = L&15.
// ---------------------------------------------------------------------------
__global__ __launch_bounds__(256) void pm_gemm_lse_kernel(
    const __bf16* __restrict__ e_hi, const __bf16* __restrict__ e_lo,
    const __bf16* __restrict__ p_hi, const __bf16* __restrict__ p_lo,
    const float* __restrict__ e2, const float* __restrict__ p2,
    const int* __restrict__ labels, float* __restrict__ row_lse) {
    __shared__ float s_e2[16];
    __shared__ int   s_lab[16];
    __shared__ float s_part[8][16];

    const int b0   = blockIdx.x * 16;
    const int tid  = threadIdx.x;
    const int wave = tid >> 5;
    const int lane = tid & 31;
    const int nrow = lane & 15;   // A row within tile == C/D column N
    const int half = lane >> 4;   // selects M offset (0 or 8) in C layout

    if (tid < 16) {
        s_e2[tid]  = e2[b0 + tid];
        s_lab[tid] = labels[b0 + tid];
    }
    __syncthreads();

    float lsum[8];
#pragma unroll
    for (int r = 0; r < 8; ++r) lsum[r] = 0.0f;

    for (int t = 0; t < 8; ++t) {
        const int c0 = (wave * 8 + t) * 16;
        v8f acc = {};
        for (int k0 = 0; k0 < D_N; k0 += 32) {
            v16bf a_h = load_a_frag(e_hi, b0 + nrow, k0, lane);
            v16bf a_l = load_a_frag(e_lo, b0 + nrow, k0, lane);
            v16bf b_h = load_b_frag(p_hi, c0 + nrow, k0, lane);
            v16bf b_l = load_b_frag(p_lo, c0 + nrow, k0, lane);
            // dot ~= hi*hi + hi*lo + lo*hi  (lo*lo ~ 2^-18 relative, dropped)
            acc = __builtin_amdgcn_wmma_f32_16x16x32_bf16(
                false, a_h, false, b_h, (short)0, acc, false, false);
            acc = __builtin_amdgcn_wmma_f32_16x16x32_bf16(
                false, a_h, false, b_l, (short)0, acc, false, false);
            acc = __builtin_amdgcn_wmma_f32_16x16x32_bf16(
                false, a_l, false, b_h, (short)0, acc, false, false);
        }
        const int c   = c0 + nrow;
        const float pc = p2[c];
#pragma unroll
        for (int r = 0; r < 8; ++r) {
            const int m  = r + half * 8;
            float sq = s_e2[m] + pc - 2.0f * acc[r];
            float d  = sqrtf(fmaxf(sq, 0.0f) + EPS_F);
            // masked positive class contributes exp(-inf) = 0
            lsum[r] += (s_lab[m] == c) ? 0.0f : expf(-d);
        }
    }

    // Reduce each row-partial across the 16 lanes of its half (bits 0..3).
#pragma unroll
    for (int r = 0; r < 8; ++r) {
        float v = lsum[r];
        v += __shfl_xor(v, 1);
        v += __shfl_xor(v, 2);
        v += __shfl_xor(v, 4);
        v += __shfl_xor(v, 8);
        if (nrow == 0) s_part[wave][r + half * 8] = v;
    }
    __syncthreads();

    // Fixed-order combine across the 8 waves -> deterministic.
    if (tid < 16) {
        float s = 0.0f;
#pragma unroll
        for (int w = 0; w < 8; ++w) s += s_part[w][tid];
        // logsumexp(-dists) = log(sum exp(-d)); no max-shift needed:
        // d in ~[20,40] => exp(-d) in [4e-18, 2e-9], safely inside fp32 range.
        row_lse[b0 + tid] = logf(s);
    }
}

// ---------------------------------------------------------------------------
// Kernel 4: deterministic final reduction.
// out = mean(pos_dist) + MARGIN + mean(row_lse)
// (reference: pos + MARGIN - neg, with neg = -mean(logsumexp))
// ---------------------------------------------------------------------------
__global__ __launch_bounds__(256) void pm_final_kernel(
    const float* __restrict__ pos_dist, const float* __restrict__ row_lse,
    float* __restrict__ out) {
    __shared__ float s1[256];
    __shared__ float s2[256];
    float a = 0.0f, b = 0.0f;
    for (int i = threadIdx.x; i < B_N; i += 256) {
        a += pos_dist[i];
        b += row_lse[i];
    }
    s1[threadIdx.x] = a;
    s2[threadIdx.x] = b;
    __syncthreads();
    for (int s = 128; s > 0; s >>= 1) {
        if (threadIdx.x < s) {
            s1[threadIdx.x] += s1[threadIdx.x + s];
            s2[threadIdx.x] += s2[threadIdx.x + s];
        }
        __syncthreads();
    }
    if (threadIdx.x == 0)
        out[0] = s1[0] / (float)B_N + MARGIN_F + s2[0] / (float)B_N;
}

// ---------------------------------------------------------------------------
// Host launcher. Workspace layout (256B-aligned offsets, ~28.4 MB total):
//   e_hi  B*D bf16   e_lo  B*D bf16   p_hi  C*D bf16   p_lo  C*D bf16
//   e2 B f32   p2 C f32   pos_dist B f32   row_lse B f32
// ---------------------------------------------------------------------------
extern "C" void kernel_launch(void* const* d_in, const int* in_sizes, int n_in,
                              void* d_out, int out_size, void* d_ws,
                              size_t ws_size, hipStream_t stream) {
    (void)in_sizes; (void)n_in; (void)out_size; (void)ws_size;
    const float* emb    = (const float*)d_in[0];  // [B, D] fp32
    const int*   labels = (const int*)d_in[1];    // [B] int32
    const float* proto  = (const float*)d_in[2];  // [C, D] fp32

    char* ws = (char*)d_ws;
    size_t off = 0;
    auto take = [&](size_t bytes) -> char* {
        char* p = ws + off;
        off = (off + bytes + 255) & ~(size_t)255;
        return p;
    };
    __bf16* e_hi = (__bf16*)take((size_t)B_N * D_N * 2);
    __bf16* e_lo = (__bf16*)take((size_t)B_N * D_N * 2);
    __bf16* p_hi = (__bf16*)take((size_t)C_N * D_N * 2);
    __bf16* p_lo = (__bf16*)take((size_t)C_N * D_N * 2);
    float* e2       = (float*)take((size_t)B_N * 4);
    float* p2       = (float*)take((size_t)C_N * 4);
    float* pos_dist = (float*)take((size_t)B_N * 4);
    float* row_lse  = (float*)take((size_t)B_N * 4);

    pm_convert_kernel<<<B_N, 256, 0, stream>>>(emb, e_hi, e_lo, e2);
    pm_convert_kernel<<<C_N, 256, 0, stream>>>(proto, p_hi, p_lo, p2);
    pm_pos_kernel<<<B_N, 256, 0, stream>>>(emb, proto, labels, pos_dist);
    pm_gemm_lse_kernel<<<B_N / 16, 256, 0, stream>>>(
        e_hi, e_lo, p_hi, p_lo, e2, p2, labels, row_lse);
    pm_final_kernel<<<1, 256, 0, stream>>>(pos_dist, row_lse, (float*)d_out);
}
